// GFEModule_39711267619166
// MI455X (gfx1250) — matrix operand order repaired and checked by reading, not compiled, not run on hardware
//
#include <hip/hip_runtime.h>
#include <hip/hip_bf16.h>

typedef __attribute__((ext_vector_type(16))) _Float16 v16h;
typedef __attribute__((ext_vector_type(8)))  _Float16 v8h;
typedef __attribute__((ext_vector_type(8)))  float    v8f;

#define NF 32
#define BN 128
#define C 256
#define S 256
#define HH 16
#define WW 16
#define NHEADS 8
#define DH 32

// ---- WMMA fragment loaders (CDNA5 ISA 7.12.2, wave32) ----
// A 16x32 f16: lane row m=lane%16; elems e<8 -> k=kb+half*8+e ; e>=8 -> k=kb+16+half*8+(e-8)
__device__ __forceinline__ v16h loadA(const _Float16* rowp, int kb, int half) {
    v8h lo = *(const v8h*)(rowp + kb + (half << 3));
    v8h hi = *(const v8h*)(rowp + kb + 16 + (half << 3));
    return __builtin_shufflevector(lo, hi, 0, 1, 2, 3, 4, 5, 6, 7, 8, 9, 10, 11, 12, 13, 14, 15);
}
// B 32x16 f16: lane col n=lane%16; elems e<8 -> k=kb+half*8+e ; e>=8 -> k=kb+16+half*8+(e-8)
__device__ __forceinline__ v16h loadB(const _Float16* rowp, int kb, int half) {
    v8h lo = *(const v8h*)(rowp + kb + (half << 3));
    v8h hi = *(const v8h*)(rowp + kb + 16 + (half << 3));
    return __builtin_shufflevector(lo, hi, 0, 1, 2, 3, 4, 5, 6, 7, 8, 9, 10, 11, 12, 13, 14, 15);
}
__device__ __forceinline__ v8f wmma16(v16h a, v16h b, v8f c) {
    return __builtin_amdgcn_wmma_f32_16x16x32_f16(false, a, false, b, (short)0, c, false, false);
}
__device__ __forceinline__ v8f vzero8() {
    v8f z;
#pragma unroll
    for (int i = 0; i < 8; ++i) z[i] = 0.0f;
    return z;
}

// ============ 0a) weight f32 -> f16 ============
__global__ __launch_bounds__(256) void w2h_kernel(const float* __restrict__ w, _Float16* __restrict__ wh, int n) {
    int i = blockIdx.x * 256 + threadIdx.x;
    if (i < n) wh[i] = (_Float16)w[i];
}

// ============ 0b) tokens: feat (BN,C,S) f32 -> tok (BN,S,C) f16, LDS-tiled transpose ============
__global__ __launch_bounds__(256) void transpose_kernel(const float* __restrict__ feat, _Float16* __restrict__ tok) {
    __shared__ float tile[16][17];
    int wg = blockIdx.x;
    int sb = wg & 15; wg >>= 4;
    int cb = wg & 15; int bn = wg >> 4;
    int tx = threadIdx.x & 15, ty = threadIdx.x >> 4;
    tile[ty][tx] = feat[(size_t)bn * C * S + (cb * 16 + ty) * S + sb * 16 + tx];
    __syncthreads();
    tok[(size_t)bn * S * C + (sb * 16 + ty) * C + cb * 16 + tx] = (_Float16)tile[tx][ty];
}

// ============ 1) QKV projection: M=s, N=out-ch, K=in-ch; windowed K/V sources ============
// q,k stored (S,C); v stored transposed (C,S) for the attn*V stage
__global__ __launch_bounds__(256) void qkv_kernel(const _Float16* __restrict__ tok,
                                                  const _Float16* __restrict__ wh,
                                                  const float* __restrict__ bio,
                                                  _Float16* __restrict__ qh,
                                                  _Float16* __restrict__ kh,
                                                  _Float16* __restrict__ vt) {
    int wg = blockIdx.x;
    int mblk = wg & 15; wg >>= 4;
    int which = wg % 3;
    int bn = wg / 3;
    int n = bn & (NF - 1), b = bn >> 5;
    int src = bn;
    if (which == 1) src = b * NF + (n > 0 ? n - 1 : 0);
    else if (which == 2) src = b * NF + (n < NF - 1 ? n + 1 : NF - 1);
    int lane = threadIdx.x & 31, wave = threadIdx.x >> 5;
    int half = lane >> 4, l16 = lane & 15;
    int s0 = mblk * 16;
    const _Float16* Arow = tok + (size_t)src * S * C + (s0 + l16) * C;
    const _Float16* Wbase = wh + (size_t)which * C * C;
    const float scale = (which == 0) ? 0.17677669529663687f : 1.0f;  // 1/sqrt(dh)
    for (int nt = 0; nt < 2; ++nt) {
        int n0 = (wave * 2 + nt) * 16;
        const _Float16* Brow = Wbase + (n0 + l16) * C;
        v8f acc = vzero8();
        for (int kt = 0; kt < C / 32; ++kt) {
            acc = wmma16(loadA(Arow, kt * 32, half), loadB(Brow, kt * 32, half), acc);
        }
        float bv = bio[which * C + n0 + l16];
#pragma unroll
        for (int r = 0; r < 8; ++r) {
            int s = s0 + r + (half << 3);
            int o = n0 + l16;
            float v = (acc[r] + bv) * scale;
            if (which == 0)      qh[(size_t)bn * S * C + s * C + o] = (_Float16)v;
            else if (which == 1) kh[(size_t)bn * S * C + s * C + o] = (_Float16)v;
            else                 vt[(size_t)bn * C * S + o * S + s] = (_Float16)v;
        }
    }
}

// ============ 2) Attention per (frame, head, 16-row q block), one wave ============
__global__ __launch_bounds__(32) void attn_kernel(const _Float16* __restrict__ qh,
                                                  const _Float16* __restrict__ kh,
                                                  const _Float16* __restrict__ vt,
                                                  _Float16* __restrict__ ctx) {
    __shared__ __align__(16) float sc[16 * 256];
    __shared__ __align__(16) _Float16 pm[16 * 256];
    int wg = blockIdx.x;
    int qb = wg & 15; wg >>= 4;
    int head = wg & 7; int bn = wg >> 3;
    int lane = threadIdx.x, half = lane >> 4, l16 = lane & 15;
    int hc = head * DH;
    const _Float16* qrow = qh + (size_t)bn * S * C + (qb * 16 + l16) * C + hc;
    const _Float16* kbase = kh + (size_t)bn * S * C;
    v16h aq = loadA(qrow, 0, half);
    for (int kt = 0; kt < 16; ++kt) {
        v16h bf = loadB(kbase + (kt * 16 + l16) * C + hc, 0, half);  // B[d][n] = K[tok n][d]
        v8f acc = vzero8();
        acc = wmma16(aq, bf, acc);  // dh == 32 == K: one WMMA per score tile
#pragma unroll
        for (int r = 0; r < 8; ++r) sc[(r + (half << 3)) * 256 + kt * 16 + l16] = acc[r];
    }
    __syncthreads();
    {   // softmax: each row split across two lanes (l16, l16+16), combined via shfl_xor
        int row = l16;
        int base = half * 128;
        float mx = -3.4e38f;
        for (int j = 0; j < 128; ++j) mx = fmaxf(mx, sc[row * 256 + base + j]);
        mx = fmaxf(mx, __shfl_xor(mx, 16, 32));
        float sum = 0.0f;
        for (int j = 0; j < 128; ++j) {
            float e = __expf(sc[row * 256 + base + j] - mx);
            sum += e;
            sc[row * 256 + base + j] = e;
        }
        sum += __shfl_xor(sum, 16, 32);
        float inv = 1.0f / sum;
        for (int j = 0; j < 128; ++j) pm[row * 256 + base + j] = (_Float16)(sc[row * 256 + base + j] * inv);
    }
    __syncthreads();
    const _Float16* vbase = vt + (size_t)bn * C * S;
    for (int nt = 0; nt < 2; ++nt) {
        v8f acc = vzero8();
        const _Float16* Brow = vbase + (hc + nt * 16 + l16) * S;  // V^T row: contiguous over tokens
        for (int kt = 0; kt < 8; ++kt) {
            acc = wmma16(loadA(pm + l16 * 256, kt * 32, half), loadB(Brow, kt * 32, half), acc);
        }
#pragma unroll
        for (int r = 0; r < 8; ++r)
            ctx[(size_t)bn * S * C + (qb * 16 + r + (half << 3)) * C + hc + nt * 16 + l16] = (_Float16)acc[r];
    }
}

// ============ 3) out_proj GEMM + bias + token residual + LayerNorm(C), fused ============
// output feat_attn fa: (BN,S,C) f32
__global__ __launch_bounds__(256) void outproj_ln_kernel(const _Float16* __restrict__ ctx,
                                                         const _Float16* __restrict__ woh,
                                                         const float* __restrict__ bo,
                                                         const float* __restrict__ feat,
                                                         const float* __restrict__ g,
                                                         const float* __restrict__ be,
                                                         float* __restrict__ fa) {
    __shared__ float xs[16 * 256];
    __shared__ float mu[16], inv[16];
    int wg = blockIdx.x;
    int mblk = wg & 15; int bn = wg >> 4;
    int s0 = mblk * 16;
    int lane = threadIdx.x & 31, wave = threadIdx.x >> 5;
    int half = lane >> 4, l16 = lane & 15;
    const _Float16* Arow = ctx + (size_t)bn * S * C + (s0 + l16) * C;
    for (int nt = 0; nt < 2; ++nt) {
        int n0 = (wave * 2 + nt) * 16;
        const _Float16* Brow = woh + (n0 + l16) * C;
        v8f acc = vzero8();
        for (int kt = 0; kt < 8; ++kt)
            acc = wmma16(loadA(Arow, kt * 32, half), loadB(Brow, kt * 32, half), acc);
        float bv = bo[n0 + l16];
#pragma unroll
        for (int r = 0; r < 8; ++r) {
            int m = r + (half << 3);
            int cc = n0 + l16;
            xs[m * 256 + cc] = acc[r] + bv + feat[(size_t)bn * C * S + cc * S + s0 + m];
        }
    }
    __syncthreads();
    if (threadIdx.x < 16) {
        int r = threadIdx.x;
        float s1 = 0.0f, s2 = 0.0f;
        for (int j = 0; j < C; ++j) { float v = xs[r * 256 + j]; s1 += v; s2 += v * v; }
        float m_ = s1 / C;
        mu[r] = m_;
        inv[r] = rsqrtf(s2 / C - m_ * m_ + 1e-5f);
    }
    __syncthreads();
    int r = threadIdx.x >> 4;
    int c0 = threadIdx.x & 15;
    for (int i = 0; i < 16; ++i) {
        int cc = c0 + i * 16;
        fa[(size_t)bn * S * C + (s0 + r) * C + cc] = (xs[r * 256 + cc] - mu[r]) * inv[r] * g[cc] + be[cc];
    }
}

// ============ 4-7) dynamic-conv small ops (all on (S,C) layout -> coalesced) ============
__global__ __launch_bounds__(256) void pool_kernel(const float* __restrict__ fa, float* __restrict__ ga) {
    int bn = blockIdx.x; int c = threadIdx.x;
    const float* p = fa + (size_t)bn * S * C + c;
    float s = 0.0f;
    for (int i = 0; i < S; ++i) s += p[i * C];
    ga[bn * C + c] = s * (1.0f / S);
}

__global__ __launch_bounds__(256) void kg1_kernel(const float* __restrict__ ga, const float* __restrict__ w,
                                                  const float* __restrict__ b, float* __restrict__ h1) {
    __shared__ float gs[C];
    int bn = blockIdx.x; int o = threadIdx.x;
    gs[o] = ga[bn * C + o];
    __syncthreads();
    float s = b[o];
    const float* wr = w + (size_t)o * C;
    for (int k = 0; k < C; ++k) s += gs[k] * wr[k];
    h1[bn * C + o] = fmaxf(s, 0.0f);
}

__global__ __launch_bounds__(256) void kg2_kernel(const float* __restrict__ h1, const float* __restrict__ w,
                                                  const float* __restrict__ b, float* __restrict__ kern) {
    __shared__ float hs[C];
    int part = blockIdx.x % 9; int bn = blockIdx.x / 9;
    int o = part * 256 + threadIdx.x;
    hs[threadIdx.x] = h1[bn * C + threadIdx.x];
    __syncthreads();
    float s = b[o];
    const float* wr = w + (size_t)o * C;
    for (int k = 0; k < C; ++k) s += hs[k] * wr[k];
    kern[bn * 2304 + o] = s;
}

__global__ __launch_bounds__(256) void dconv_kernel(const float* __restrict__ fa, const float* __restrict__ kern,
                                                    _Float16* __restrict__ outh) {
    int s = blockIdx.x & (S - 1); int bn = blockIdx.x >> 8;
    int h = s >> 4, w = s & 15;
    int c = threadIdx.x;
    const float* kp = kern + bn * 2304 + c * 9;
    const float* xp = fa + (size_t)bn * S * C + c;
    float sum = 0.0f;
#pragma unroll
    for (int i = 0; i < 3; ++i) {
        int h2 = h + i - 1;
        if (h2 < 0 || h2 >= HH) continue;
#pragma unroll
        for (int j = 0; j < 3; ++j) {
            int w2 = w + j - 1;
            if (w2 < 0 || w2 >= WW) continue;
            sum += kp[i * 3 + j] * xp[(h2 * WW + w2) * C];
        }
    }
    outh[(size_t)bn * S * C + s * C + c] = (_Float16)sum;
}

// ============ 8) 'oc' GEMMs computed transposed: out^T[s][o] = sum_c X^T[s][c] W[o][c] ============
// mode 0: dc_proj -> f16 (S,C) ; mode 1: fc1 -> relu f16 (S,C) ; mode 2: fc2 -> f32 (S,C) + resid
__global__ __launch_bounds__(256) void gemm_sc_kernel(const _Float16* __restrict__ Wh,
                                                      const float* __restrict__ bias,
                                                      const _Float16* __restrict__ Xh,
                                                      float* __restrict__ outf,
                                                      _Float16* __restrict__ outh,
                                                      const float* __restrict__ resid,
                                                      int mode) {
    int wg = blockIdx.x;
    int mblk = wg & 15; int bn = wg >> 4;
    int s0 = mblk * 16;
    int lane = threadIdx.x & 31, wave = threadIdx.x >> 5;
    int half = lane >> 4, l16 = lane & 15;
    const _Float16* Arow = Xh + (size_t)bn * S * C + (s0 + l16) * C;
    for (int nt = 0; nt < 2; ++nt) {
        int n0 = (wave * 2 + nt) * 16;
        const _Float16* Brow = Wh + (n0 + l16) * C;
        v8f acc = vzero8();
        for (int kt = 0; kt < 8; ++kt)
            acc = wmma16(loadA(Arow, kt * 32, half), loadB(Brow, kt * 32, half), acc);
        float bv = bias[n0 + l16];
#pragma unroll
        for (int r = 0; r < 8; ++r) {
            int s = s0 + r + (half << 3);
            int o = n0 + l16;
            float v = acc[r] + bv;
            size_t oi = (size_t)bn * S * C + s * C + o;
            if (mode == 0) outh[oi] = (_Float16)v;
            else if (mode == 1) outh[oi] = (_Float16)fmaxf(v, 0.0f);
            else outf[oi] = v + resid[oi];
        }
    }
}

// ============ 9) GroupNorm(32): 8 contiguous channels per thread via one b128 load ============
__global__ __launch_bounds__(256) void groupnorm_kernel(const _Float16* __restrict__ xh,
                                                        const float* __restrict__ g,
                                                        const float* __restrict__ b,
                                                        _Float16* __restrict__ outh) {
    __shared__ float rs1[256], rs2[256];
    int bn = blockIdx.x >> 5;
    int grp = blockIdx.x & 31;
    int t = threadIdx.x;  // spatial index
    v8h xv = *(const v8h*)(xh + (size_t)bn * S * C + t * C + grp * 8);
    float vals[8];
    float s1 = 0.0f, s2 = 0.0f;
#pragma unroll
    for (int i = 0; i < 8; ++i) { float v = (float)xv[i]; vals[i] = v; s1 += v; s2 += v * v; }
    rs1[t] = s1; rs2[t] = s2;
    __syncthreads();
    for (int off = 128; off > 0; off >>= 1) {
        if (t < off) { rs1[t] += rs1[t + off]; rs2[t] += rs2[t + off]; }
        __syncthreads();
    }
    float m_ = rs1[0] / 2048.0f;
    float inv = rsqrtf(rs2[0] / 2048.0f - m_ * m_ + 1e-5f);
    v8h ov;
#pragma unroll
    for (int i = 0; i < 8; ++i) {
        int c = grp * 8 + i;
        ov[i] = (_Float16)((vals[i] - m_) * inv * g[c] + b[c]);
    }
    *(v8h*)(outh + (size_t)bn * S * C + t * C + grp * 8) = ov;
}

// ============ 10) final per-sample LayerNorm([C,H,W]); y is (S,C), output (C,S) ============
__global__ __launch_bounds__(256) void finalln_kernel(const float* __restrict__ y,
                                                      const float* __restrict__ g,
                                                      const float* __restrict__ b,
                                                      float* __restrict__ out) {
    __shared__ float rs1[256], rs2[256];
    int bn = blockIdx.x;
    int t = threadIdx.x;
    const float* yp = y + (size_t)bn * S * C;
    float s1 = 0.0f, s2 = 0.0f;
    for (int i = 0; i < 256; ++i) { float v = yp[i * C + t]; s1 += v; s2 += v * v; }
    rs1[t] = s1; rs2[t] = s2;
    __syncthreads();
    for (int off = 128; off > 0; off >>= 1) {
        if (t < off) { rs1[t] += rs1[t + off]; rs2[t] += rs2[t + off]; }
        __syncthreads();
    }
    float m_ = rs1[0] / 65536.0f;
    float inv = rsqrtf(rs2[0] / 65536.0f - m_ * m_ + 1e-5f);
    float* op = out + (size_t)bn * C * S;
    // thread t owns channel c=t; loop spatial: y[s*C+c] coalesced reads, per-thread contiguous writes
    for (int i = 0; i < 256; ++i) {
        int idx = t * S + i;  // (c,s) flat = gamma/beta/output index
        op[idx] = (yp[i * C + t] - m_) * inv * g[idx] + b[idx];
    }
}

extern "C" void kernel_launch(void* const* d_in, const int* in_sizes, int n_in,
                              void* d_out, int out_size, void* d_ws, size_t ws_size,
                              hipStream_t stream) {
    (void)in_sizes; (void)n_in; (void)out_size; (void)ws_size;
    const float* feat       = (const float*)d_in[0];
    const float* in_proj_w  = (const float*)d_in[1];
    const float* in_proj_b  = (const float*)d_in[2];
    const float* out_proj_w = (const float*)d_in[3];
    const float* out_proj_b = (const float*)d_in[4];
    const float* ln1_g      = (const float*)d_in[5];
    const float* ln1_b      = (const float*)d_in[6];
    const float* kg_w1      = (const float*)d_in[7];
    const float* kg_b1      = (const float*)d_in[8];
    const float* kg_w2      = (const float*)d_in[9];
    const float* kg_b2      = (const float*)d_in[10];
    const float* dc_proj_w  = (const float*)d_in[11];
    const float* dc_proj_b  = (const float*)d_in[12];
    const float* gn_g       = (const float*)d_in[13];
    const float* gn_b       = (const float*)d_in[14];
    const float* fc1_w      = (const float*)d_in[15];
    const float* fc1_b      = (const float*)d_in[16];
    const float* fc2_w      = (const float*)d_in[17];
    const float* fc2_b      = (const float*)d_in[18];
    const float* dcn_g      = (const float*)d_in[19];
    const float* dcn_b      = (const float*)d_in[20];

    char* base = (char*)d_ws;
    const size_t A = (size_t)BN * S * C;  // 8,388,608 elements
    _Float16* tok  = (_Float16*)base;     // (BN,S,C) f16 tokens
    _Float16* qh   = tok + A;
    _Float16* kh   = qh + A;
    _Float16* vt   = kh + A;              // V transposed: (BN,C,S)
    _Float16* ctxh = vt + A;
    float* fa   = (float*)(base + 10 * A);   // feat_attn (BN,S,C) f32
    float* ybuf = fa + A;                    // fc2 output (BN,S,C) f32
    float* ga   = ybuf + A;
    float* h1   = ga + BN * C;
    float* kern = h1 + BN * C;
    _Float16* wqkv_h = (_Float16*)(kern + BN * 2304);  // 7*C*C f16 weights
    _Float16* wout_h = wqkv_h + 3 * C * C;
    _Float16* wdc_h  = wqkv_h + 4 * C * C;
    _Float16* wfc1_h = wqkv_h + 5 * C * C;
    _Float16* wfc2_h = wqkv_h + 6 * C * C;
    // reuse attention buffers (dead after outproj) for the conv/FC chain
    _Float16* convh = tok;   // depthwise conv out (S,C) f16
    _Float16* dcph  = qh;    // dc_proj out (S,C) f16, pre-GN
    _Float16* dch   = kh;    // GroupNorm out (S,C) f16
    _Float16* hfc1  = vt;    // fc1 out (S,C) f16

    w2h_kernel<<<(3 * C * C) / 256, 256, 0, stream>>>(in_proj_w, wqkv_h, 3 * C * C);
    w2h_kernel<<<(C * C) / 256, 256, 0, stream>>>(out_proj_w, wout_h, C * C);
    w2h_kernel<<<(C * C) / 256, 256, 0, stream>>>(dc_proj_w, wdc_h, C * C);
    w2h_kernel<<<(C * C) / 256, 256, 0, stream>>>(fc1_w, wfc1_h, C * C);
    w2h_kernel<<<(C * C) / 256, 256, 0, stream>>>(fc2_w, wfc2_h, C * C);
    transpose_kernel<<<BN * 256, 256, 0, stream>>>(feat, tok);

    qkv_kernel<<<BN * 3 * 16, 256, 0, stream>>>(tok, wqkv_h, in_proj_b, qh, kh, vt);
    attn_kernel<<<BN * NHEADS * 16, 32, 0, stream>>>(qh, kh, vt, ctxh);
    outproj_ln_kernel<<<BN * 16, 256, 0, stream>>>(ctxh, wout_h, out_proj_b, feat, ln1_g, ln1_b, fa);

    pool_kernel<<<BN, 256, 0, stream>>>(fa, ga);
    kg1_kernel<<<BN, 256, 0, stream>>>(ga, kg_w1, kg_b1, h1);
    kg2_kernel<<<BN * 9, 256, 0, stream>>>(h1, kg_w2, kg_b2, kern);
    dconv_kernel<<<BN * S, 256, 0, stream>>>(fa, kern, convh);

    gemm_sc_kernel<<<BN * 16, 256, 0, stream>>>(wdc_h, dc_proj_b, convh, nullptr, dcph, nullptr, 0);
    groupnorm_kernel<<<BN * 32, 256, 0, stream>>>(dcph, gn_g, gn_b, dch);
    gemm_sc_kernel<<<BN * 16, 256, 0, stream>>>(wfc1_h, fc1_b, dch, nullptr, hfc1, nullptr, 1);
    gemm_sc_kernel<<<BN * 16, 256, 0, stream>>>(wfc2_h, fc2_b, hfc1, ybuf, nullptr, fa, 2);
    finalln_kernel<<<BN, 256, 0, stream>>>(ybuf, dcn_g, dcn_b, (float*)d_out);
}
